// CrossAttention_60825326846278
// MI455X (gfx1250) — matrix operand order repaired and checked by reading, not compiled, not run on hardware
//
#include <hip/hip_runtime.h>
#include <stdint.h>

typedef __attribute__((ext_vector_type(16))) _Float16 v16h;
typedef __attribute__((ext_vector_type(8)))  float    v8f;
typedef __attribute__((ext_vector_type(2)))  _Float16 h2;

#define DEVINL __device__ __forceinline__

static DEVINL v8f wmma_f16(v16h a, v16h b, v8f c) {
  // D = A(16x32 f16) * B(32x16 f16) + C(16x16 f32)
  return __builtin_amdgcn_wmma_f32_16x16x32_f16(false, a, false, b, (short)0, c,
                                                false, false);
}

// ---- Fragment loaders (contiguous k order, k = (lane>>4)*16 + s, wave32) ----
// A fragment 16x32: src -> (row0,k0), row stride ldr (elems).
// lane&15 = row; VGPR j -> K = (lane>>4)*16 + 2j.
static DEVINL v16h load_frag_a(const _Float16* src, int ldr, int lane) {
  const int row = lane & 15;
  const int kh  = (lane >> 4) * 16;
  v16h a;
#pragma unroll
  for (int j = 0; j < 8; ++j) {
    const int kb = kh + 2 * j;
    h2 p = *(const h2*)(src + (size_t)row * ldr + kb);
    a[2 * j]     = p.x;
    a[2 * j + 1] = p.y;
  }
  return a;
}

// B fragment 32x16 (KxN) from column-major storage srcT[col][k], stride ldc.
// lane&15 = col; VGPR j -> K = (lane>>4)*16 + 2j (pair contiguous in srcT).
static DEVINL v16h load_frag_b_T(const _Float16* srcT, int ldc, int lane) {
  const int col = lane & 15;
  const int kh  = (lane >> 4) * 16;
  v16h b;
#pragma unroll
  for (int j = 0; j < 8; ++j) {
    h2 p = *(const h2*)(srcT + (size_t)col * ldc + kh + 2 * j);
    b[2 * j]     = p.x;
    b[2 * j + 1] = p.y;
  }
  return b;
}

// B fragment 32x16 (KxN) from row-major storage src[k][n], stride lds.
static DEVINL v16h load_frag_b_rows(const _Float16* src, int lds, int lane) {
  const int col = lane & 15;
  const int kh  = (lane >> 4) * 16;
  v16h b;
#pragma unroll
  for (int j = 0; j < 8; ++j) {
    b[2 * j]     = src[(size_t)(kh + 2 * j) * lds + col];
    b[2 * j + 1] = src[(size_t)(kh + 2 * j + 1) * lds + col];
  }
  return b;
}

// ================= GEMM: Y[R,1024] = A[R,1024] @ W[1024,1024] (+bias) =======
// Block tile 128x64, 8 waves, each wave computes 32x32 (4 WMMA tiles).
constexpr int KDIM = 1024;
constexpr int ASTR = 34;  // 32 + pad
constexpr int BSTR = 34;

template <bool A_F16, bool OUT_F16, bool BIAS>
__global__ __launch_bounds__(256) void gemm_wmma_kernel(
    const void* __restrict__ Ap, const float* __restrict__ Wp,
    const float* __restrict__ bias, void* __restrict__ Yp) {
  __shared__ _Float16 As[128 * ASTR];  // [row][k]
  __shared__ _Float16 Bt[64 * BSTR];   // [col][k] (transposed W panel)

  const int t    = threadIdx.x;
  const int lane = t & 31;
  const int w    = t >> 5;
  const int c0   = blockIdx.x * 64;
  const int r0   = blockIdx.y * 128;
  const int cb   = (w & 1) * 32;   // wave's 32-col patch within 64
  const int rb   = (w >> 1) * 32;  // wave's 32-row patch within 128

  const float*    Af = (const float*)Ap;
  const _Float16* Ah = (const _Float16*)Ap;

  v8f acc00 = {}, acc01 = {}, acc10 = {}, acc11 = {};

  for (int k0 = 0; k0 < KDIM; k0 += 32) {
    __syncthreads();
#pragma unroll
    for (int i = 0; i < 16; ++i) {  // A panel 128x32
      const int lin = i * 256 + t;
      const int r = lin >> 5, k = lin & 31;
      const size_t gi = (size_t)(r0 + r) * KDIM + (k0 + k);
      if constexpr (A_F16) As[r * ASTR + k] = Ah[gi];
      else                 As[r * ASTR + k] = (_Float16)Af[gi];
    }
#pragma unroll
    for (int i = 0; i < 8; ++i) {   // W panel 32x64 -> transposed
      const int lin = i * 256 + t;
      const int kk = lin >> 6, c = lin & 63;
      Bt[c * BSTR + kk] = (_Float16)Wp[(size_t)(k0 + kk) * 1024 + (c0 + c)];
    }
    __syncthreads();

    // Prefetch next K panels into cache while WMMAs run.
    if (k0 + 32 < KDIM) {
      {  // A: 128 rows x 128B -> 256 cachelines, 1/thread
        const int r = t >> 1, half = t & 1;
        const size_t gi = (size_t)(r0 + r) * KDIM + (k0 + 32 + half * 16);
        if constexpr (A_F16) __builtin_prefetch(Ah + gi, 0, 3);
        else                 __builtin_prefetch(Af + gi, 0, 3);
      }
      if (t < 128) {  // W: 32 rows x 256B -> 128 cachelines
        const int kk = t >> 2, q = t & 3;
        __builtin_prefetch(Wp + (size_t)(k0 + 32 + kk) * 1024 + c0 + q * 16, 0, 3);
      }
    }

    const v16h a0 = load_frag_a(As + rb * ASTR, ASTR, lane);
    const v16h a1 = load_frag_a(As + (rb + 16) * ASTR, ASTR, lane);
    const v16h b0 = load_frag_b_T(Bt + cb * BSTR, BSTR, lane);
    const v16h b1 = load_frag_b_T(Bt + (cb + 16) * BSTR, BSTR, lane);
    acc00 = wmma_f16(a0, b0, acc00);
    acc01 = wmma_f16(a0, b1, acc01);
    acc10 = wmma_f16(a1, b0, acc10);
    acc11 = wmma_f16(a1, b1, acc11);
  }

  const int col0  = c0 + cb + (lane & 15);
  const int rbase = r0 + rb + (lane >> 4) * 8;
  float bv0 = 0.0f, bv1 = 0.0f;
  if constexpr (BIAS) { bv0 = bias[col0]; bv1 = bias[col0 + 16]; }
#pragma unroll
  for (int j = 0; j < 8; ++j) {
    const size_t i00 = (size_t)(rbase + j) * 1024 + col0;
    const size_t i10 = (size_t)(rbase + 16 + j) * 1024 + col0;
    if constexpr (OUT_F16) {
      ((_Float16*)Yp)[i00]      = (_Float16)acc00[j];
      ((_Float16*)Yp)[i00 + 16] = (_Float16)acc01[j];
      ((_Float16*)Yp)[i10]      = (_Float16)acc10[j];
      ((_Float16*)Yp)[i10 + 16] = (_Float16)acc11[j];
    } else {
      ((float*)Yp)[i00]      = acc00[j] + bv0;
      ((float*)Yp)[i00 + 16] = acc01[j] + bv1;
      ((float*)Yp)[i10]      = acc10[j] + bv0;
      ((float*)Yp)[i10 + 16] = acc11[j] + bv1;
    }
  }
}

// ================= Flash attention =========================================
constexpr int AN   = 2048;  // queries
constexpr int AM   = 2048;  // keys
constexpr int HD   = 64;
constexpr int INN  = 1024;
constexpr int KSTR = 72;    // [key][d] row stride (144B, 16B-aligned)
constexpr int PSTR = 40;    // per-wave P tile [row][key32]

__global__ __launch_bounds__(256) void attn_fwd_kernel(
    const _Float16* __restrict__ Qh, const _Float16* __restrict__ Kh,
    const _Float16* __restrict__ Vh, const unsigned char* __restrict__ mask,
    _Float16* __restrict__ Oh) {
  __shared__ _Float16 Ks[32 * KSTR];       // keys x 64 dims
  __shared__ _Float16 Vs[32 * KSTR];       // keys x 64 dims
  __shared__ _Float16 Pst[8 * 16 * PSTR];  // per-wave P tile

  const int t = threadIdx.x, lane = t & 31, w = t >> 5;
  const int b = blockIdx.z, h = blockIdx.y;
  const int qbase = blockIdx.x * 128 + w * 16;

  // Q fragments (resident whole kernel): d 0..31 and 32..63
  const size_t qoff = ((size_t)b * AN + qbase) * INN + h * HD;
  const v16h qa0 = load_frag_a(Qh + qoff, INN, lane);
  const v16h qa1 = load_frag_a(Qh + qoff + 32, INN, lane);

  // Async-DMA bases / offsets (GVS addressing: saddr64 + voffset32)
  const uint64_t Kbase = (uint64_t)(Kh + ((size_t)b * AM) * INN + h * HD);
  const uint64_t Vbase = (uint64_t)(Vh + ((size_t)b * AM) * INN + h * HD);
  const int skey = t >> 3;            // 0..31
  const int sd   = (t & 7) * 8;       // 0..56, 16B chunks
  const uint32_t ldsK = (uint32_t)(uintptr_t)&Ks[skey * KSTR + sd];
  const uint32_t ldsV = (uint32_t)(uintptr_t)&Vs[skey * KSTR + sd];

  v8f o0 = {}, o1 = {}, o2 = {}, o3 = {};
  float mrow[8], lrow[8];
#pragma unroll
  for (int j = 0; j < 8; ++j) { mrow[j] = -3.4e38f; lrow[j] = 0.0f; }

  const float scale = 0.125f;  // 1/sqrt(64)
  _Float16* Pw = Pst + w * 16 * PSTR;

  for (int kb = 0; kb < AM; kb += 32) {
    __syncthreads();
    // Stage K/V panels via CDNA5 async global->LDS DMA (ASYNCcnt tracked).
    {
      const uint32_t voff = (uint32_t)(((kb + skey) * INN + sd) * 2);
      asm volatile("global_load_async_to_lds_b128 %0, %1, %2"
                   :: "v"(ldsK), "v"(voff), "s"(Kbase) : "memory");
      asm volatile("global_load_async_to_lds_b128 %0, %1, %2"
                   :: "v"(ldsV), "v"(voff), "s"(Vbase) : "memory");
      asm volatile("s_wait_asynccnt 0x0" ::: "memory");
    }
    __syncthreads();

    // S = Q K^T for 32 keys: two 16x16 C tiles (key groups), K-dim = 64 dims
    v8f s0 = {}, s1 = {};
    s0 = wmma_f16(qa0, load_frag_b_T(Ks, KSTR, lane), s0);
    s0 = wmma_f16(qa1, load_frag_b_T(Ks + 32, KSTR, lane), s0);
    s1 = wmma_f16(qa0, load_frag_b_T(Ks + 16 * KSTR, KSTR, lane), s1);
    s1 = wmma_f16(qa1, load_frag_b_T(Ks + 16 * KSTR + 32, KSTR, lane), s1);

    const int keyc = kb + (lane & 15);
    const bool mk0 = mask[(size_t)b * AM + keyc] != 0;
    const bool mk1 = mask[(size_t)b * AM + keyc + 16] != 0;

    // Online softmax; row r = (lane>>4)*8 + j lives across 16 lanes of a half.
    float alpha[8];
#pragma unroll
    for (int j = 0; j < 8; ++j) {
      float v0 = mk0 ? s0[j] * scale : -3.0e38f;
      float v1 = mk1 ? s1[j] * scale : -3.0e38f;
      float rm = fmaxf(v0, v1);
      rm = fmaxf(rm, __shfl_xor(rm, 1));
      rm = fmaxf(rm, __shfl_xor(rm, 2));
      rm = fmaxf(rm, __shfl_xor(rm, 4));
      rm = fmaxf(rm, __shfl_xor(rm, 8));
      const float nm = fmaxf(mrow[j], rm);
      const float al = __expf(mrow[j] - nm);
      const float p0 = __expf(v0 - nm);
      const float p1 = __expf(v1 - nm);
      float ps = p0 + p1;
      ps += __shfl_xor(ps, 1);
      ps += __shfl_xor(ps, 2);
      ps += __shfl_xor(ps, 4);
      ps += __shfl_xor(ps, 8);
      lrow[j]  = lrow[j] * al + ps;
      mrow[j]  = nm;
      alpha[j] = al;
      s0[j] = p0;
      s1[j] = p1;
    }
#pragma unroll
    for (int j = 0; j < 8; ++j) {
      o0[j] *= alpha[j]; o1[j] *= alpha[j];
      o2[j] *= alpha[j]; o3[j] *= alpha[j];
    }

    // C-layout P (f32) -> f16 A-fragment via per-wave LDS patch
    const int prow = (lane >> 4) * 8, pcol = lane & 15;
#pragma unroll
    for (int j = 0; j < 8; ++j) {
      Pw[(prow + j) * PSTR + pcol]      = (_Float16)s0[j];
      Pw[(prow + j) * PSTR + 16 + pcol] = (_Float16)s1[j];
    }
    asm volatile("s_wait_dscnt 0x0" ::: "memory");  // intra-wave DS ordering

    const v16h pa = load_frag_a(Pw, PSTR, lane);
    o0 = wmma_f16(pa, load_frag_b_rows(Vs, KSTR, lane), o0);
    o1 = wmma_f16(pa, load_frag_b_rows(Vs + 16, KSTR, lane), o1);
    o2 = wmma_f16(pa, load_frag_b_rows(Vs + 32, KSTR, lane), o2);
    o3 = wmma_f16(pa, load_frag_b_rows(Vs + 48, KSTR, lane), o3);
  }

  // Normalize and store f16 attention output
#pragma unroll
  for (int j = 0; j < 8; ++j) {
    const float inv = lrow[j] > 0.0f ? 1.0f / lrow[j] : 0.0f;
    const int qr = qbase + (lane >> 4) * 8 + j;
    const size_t base = ((size_t)b * AN + qr) * INN + h * HD + (lane & 15);
    Oh[base + 0]  = (_Float16)(o0[j] * inv);
    Oh[base + 16] = (_Float16)(o1[j] * inv);
    Oh[base + 32] = (_Float16)(o2[j] * inv);
    Oh[base + 48] = (_Float16)(o3[j] * inv);
  }
}

// ================= Launch ==================================================
extern "C" void kernel_launch(void* const* d_in, const int* in_sizes, int n_in,
                              void* d_out, int out_size, void* d_ws,
                              size_t ws_size, hipStream_t stream) {
  const float* x    = (const float*)d_in[0];
  const float* ctx  = (const float*)d_in[1];
  const unsigned char* mask = (const unsigned char*)d_in[2];
  const float* Wq = (const float*)d_in[3];
  const float* Wk = (const float*)d_in[4];
  const float* Wv = (const float*)d_in[5];
  const float* Wo = (const float*)d_in[6];
  const float* bo = (const float*)d_in[7];
  float* out = (float*)d_out;

  const size_t elems = (size_t)4 * 2048 * 1024;  // tokens * INNER per buffer
  _Float16* Qh = (_Float16*)d_ws;
  _Float16* Kh = Qh + elems;
  _Float16* Vh = Kh + elems;
  _Float16* Oh = Vh + elems;

  dim3 blk(256);
  dim3 gg(16, 64);  // 1024/64 col tiles, 8192/128 row tiles

  hipLaunchKernelGGL((gemm_wmma_kernel<false, true, false>), gg, blk, 0, stream,
                     (const void*)x, Wq, (const float*)nullptr, (void*)Qh);
  hipLaunchKernelGGL((gemm_wmma_kernel<false, true, false>), gg, blk, 0, stream,
                     (const void*)ctx, Wk, (const float*)nullptr, (void*)Kh);
  hipLaunchKernelGGL((gemm_wmma_kernel<false, true, false>), gg, blk, 0, stream,
                     (const void*)ctx, Wv, (const float*)nullptr, (void*)Vh);
  hipLaunchKernelGGL(attn_fwd_kernel, dim3(16, 16, 4), blk, 0, stream,
                     Qh, Kh, Vh, mask, Oh);
  hipLaunchKernelGGL((gemm_wmma_kernel<true, false, true>), gg, blk, 0, stream,
                     (const void*)Oh, Wo, bo, (void*)out);
}
